// GINEConvLayer_20590073217126
// MI455X (gfx1250) — matrix operand order repaired and checked by reading, not compiled, not run on hardware
//
#include <hip/hip_runtime.h>

#define D_FEAT 128
#define BN_EPS 1e-5f

typedef __attribute__((ext_vector_type(16))) __bf16 v16bf;
typedef __attribute__((ext_vector_type(8)))  float  v8f;

// ---------- helpers ----------

__device__ __forceinline__ unsigned short f2bf(float f) {
    union { float f; unsigned int u; } v; v.f = f;
    unsigned int u = v.u;
    u += 0x7FFFu + ((u >> 16) & 1u);   // round-to-nearest-even
    return (unsigned short)(u >> 16);
}

// A fragment: 16x32 bf16. Lane l: row M = l&15, half = l>>4.
// VGPR i holds K = (i<4 ? 2i : 16+2(i-4)) + 8*half  (+32*ks), pairs contiguous.
__device__ __forceinline__ v16bf frag_a(const unsigned short* rowPtr, int half, int ks) {
    union { v16bf v; unsigned int u[8]; } f;
#pragma unroll
    for (int i = 0; i < 8; ++i) {
        int k = ((i < 4) ? (2 * i) : (16 + 2 * (i - 4))) + 8 * half + 32 * ks;
        f.u[i] = *(const unsigned int*)(rowPtr + k);
    }
    return f.v;
}

// B fragment: 32x16 bf16, weights stored transposed in LDS as [n][k].
// Lane l: col N = l&15, half = l>>4. VGPR i holds K = (i<4 ? 2i : 16+2(i-4)) + 8*half (+32*ks).
__device__ __forceinline__ v16bf frag_b(const unsigned short* rowPtr, int half, int ks) {
    union { v16bf v; unsigned int u[8]; } f;
#pragma unroll
    for (int i = 0; i < 8; ++i) {
        int k = ((i < 4) ? (2 * i) : (16 + 2 * (i - 4))) + 8 * half + 32 * ks;
        f.u[i] = *(const unsigned int*)(rowPtr + k);
    }
    return f.v;
}

// ---------- kernels ----------

__global__ void k_zero(float* __restrict__ p, size_t n) {
    size_t i = (size_t)blockIdx.x * blockDim.x + threadIdx.x;
    size_t stride = (size_t)gridDim.x * blockDim.x;
    for (; i < n; i += stride) p[i] = 0.f;
}

// Edge stage: one wave per edge (lane handles 4 feats). L2-resident gather + f32 atomics.
__global__ __launch_bounds__(256) void k_edges(const float* __restrict__ x,
                                               const float* __restrict__ ea,
                                               const int* __restrict__ ei,
                                               float* __restrict__ agg, int nEdges) {
    int gw   = (int)((blockIdx.x * blockDim.x + threadIdx.x) >> 5);
    int lane = threadIdx.x & 31;
    int nw   = (int)((gridDim.x * blockDim.x) >> 5);
    const int* src = ei;
    const int* dst = ei + nEdges;
    for (int e = gw; e < nEdges; e += nw) {
        int sN = src[e], dN = dst[e];
        const float4 xv = *(const float4*)(x + (size_t)sN * D_FEAT + lane * 4);
        const float4 ev = *(const float4*)(ea + (size_t)e * D_FEAT + lane * 4);
        float4 m;
        m.x = fmaxf(xv.x + ev.x, 0.f);
        m.y = fmaxf(xv.y + ev.y, 0.f);
        m.z = fmaxf(xv.z + ev.z, 0.f);
        m.w = fmaxf(xv.w + ev.w, 0.f);
        float* ap = agg + (size_t)dN * D_FEAT + lane * 4;
        atomicAdd(ap + 0, m.x);
        atomicAdd(ap + 1, m.y);
        atomicAdd(ap + 2, m.z);
        atomicAdd(ap + 3, m.w);
    }
}

// GIN MLP: r = x + relu( relu((x+agg)@W1 + b1) @ W2 + b2 ), plus BN1 stat partials.
// agg and r may alias (agg fully consumed into LDS before r stores).
__global__ __launch_bounds__(256) void k_gin(const float* __restrict__ x,
                                             const float* agg,
                                             const float* __restrict__ w1, const float* __restrict__ b1,
                                             const float* __restrict__ w2, const float* __restrict__ b2,
                                             float* r,
                                             float* __restrict__ gsum, float* __restrict__ gsumsq,
                                             int nNodes) {
    __shared__ unsigned short shA[128 * 132];
    __shared__ unsigned short shW1[128 * 132];
    __shared__ unsigned short shW2[128 * 132];
    __shared__ unsigned short shH[128 * 132];
    __shared__ float shSum[128];
    __shared__ float shSumSq[128];

    const int tid  = threadIdx.x;
    const int wave = tid >> 5, lane = tid & 31;
    const int half = lane >> 4, lm = lane & 15;
    const int m0   = blockIdx.x * 128;

    for (int i = tid; i < 128 * 128; i += 256) {
        int k = i >> 7, n = i & 127;
        shW1[n * 132 + k] = f2bf(w1[i]);   // transpose: LDS holds W^T [n][k]
        shW2[n * 132 + k] = f2bf(w2[i]);
        int node = m0 + k;                 // reuse: k as row, n as col of A tile
        float v = 0.f;
        if (node < nNodes) v = x[(size_t)node * D_FEAT + n] + agg[(size_t)node * D_FEAT + n];
        shA[k * 132 + n] = f2bf(v);
    }
    if (tid < 128) { shSum[tid] = 0.f; shSumSq[tid] = 0.f; }
    __syncthreads();

    const v8f z8 = {0.f, 0.f, 0.f, 0.f, 0.f, 0.f, 0.f, 0.f};

    // GEMM1: H = relu(A @ W1 + b1)
    {
        v8f acc[8];
#pragma unroll
        for (int t = 0; t < 8; ++t) acc[t] = z8;
#pragma unroll
        for (int ks = 0; ks < 4; ++ks) {
            v16bf a = frag_a(shA + (wave * 16 + lm) * 132, half, ks);
#pragma unroll
            for (int t = 0; t < 8; ++t) {
                v16bf b = frag_b(shW1 + (t * 16 + lm) * 132, half, ks);
                acc[t] = __builtin_amdgcn_wmma_f32_16x16x32_bf16(
                    false, a, false, b, (short)0, acc[t], false, false);
            }
        }
#pragma unroll
        for (int t = 0; t < 8; ++t) {
            int n = t * 16 + lm;
            float bias = b1[n];
#pragma unroll
            for (int i = 0; i < 8; ++i) {
                int m = wave * 16 + i + 8 * half;
                float v = fmaxf(acc[t][i] + bias, 0.f);
                shH[m * 132 + n] = f2bf(v);
            }
        }
    }
    __syncthreads();

    // GEMM2: r = x + relu(H @ W2 + b2); accumulate BN1 partial sums
    {
        v8f acc[8];
#pragma unroll
        for (int t = 0; t < 8; ++t) acc[t] = z8;
#pragma unroll
        for (int ks = 0; ks < 4; ++ks) {
            v16bf a = frag_a(shH + (wave * 16 + lm) * 132, half, ks);
#pragma unroll
            for (int t = 0; t < 8; ++t) {
                v16bf b = frag_b(shW2 + (t * 16 + lm) * 132, half, ks);
                acc[t] = __builtin_amdgcn_wmma_f32_16x16x32_bf16(
                    false, a, false, b, (short)0, acc[t], false, false);
            }
        }
#pragma unroll
        for (int t = 0; t < 8; ++t) {
            int n = t * 16 + lm;
            float bias = b2[n];
            float ls = 0.f, ls2 = 0.f;
#pragma unroll
            for (int i = 0; i < 8; ++i) {
                int m = wave * 16 + i + 8 * half;
                int node = m0 + m;
                if (node < nNodes) {
                    float v  = fmaxf(acc[t][i] + bias, 0.f);
                    float rv = x[(size_t)node * D_FEAT + n] + v;
                    r[(size_t)node * D_FEAT + n] = rv;
                    ls  += rv;
                    ls2 += rv * rv;
                }
            }
            ls  += __shfl_xor(ls, 16);
            ls2 += __shfl_xor(ls2, 16);
            if (half == 0) {
                atomicAdd(&shSum[n], ls);
                atomicAdd(&shSumSq[n], ls2);
            }
        }
    }
    __syncthreads();
    if (tid < 128) {
        atomicAdd(&gsum[tid], shSum[tid]);
        atomicAdd(&gsumsq[tid], shSumSq[tid]);
    }
}

__global__ void k_stats(const float* __restrict__ sum, const float* __restrict__ sumsq,
                        float* __restrict__ mean, float* __restrict__ rstd, float inv_n) {
    int c = threadIdx.x;
    if (c < 128) {
        float m = sum[c] * inv_n;
        float v = sumsq[c] * inv_n - m * m;
        mean[c] = m;
        rstd[c] = rsqrtf(v + BN_EPS);
    }
}

// FFN: h = BN1(r); out = h + relu(h@FW1 + fb1)@FW2 + fb2; accumulate BN2 partials.
__global__ __launch_bounds__(256) void k_ffn(const float* __restrict__ r,
                                             const float* __restrict__ mean1, const float* __restrict__ rstd1,
                                             const float* __restrict__ g1, const float* __restrict__ bt1,
                                             const float* __restrict__ fw1, const float* __restrict__ fb1,
                                             const float* __restrict__ fw2, const float* __restrict__ fb2,
                                             float* __restrict__ outp,
                                             float* __restrict__ gsum, float* __restrict__ gsumsq,
                                             int nNodes) {
    __shared__ unsigned short shH[128 * 132];    // BN1-applied tile (bf16)
    __shared__ unsigned short shW1[256 * 132];   // FW1^T [n][k], n<256, k<128
    __shared__ unsigned short shW2[128 * 260];   // FW2^T [n][k], n<128, k<256
    __shared__ unsigned short shF[128 * 260];    // hidden activations (bf16)
    __shared__ float shSum[128];
    __shared__ float shSumSq[128];

    const int tid  = threadIdx.x;
    const int wave = tid >> 5, lane = tid & 31;
    const int half = lane >> 4, lm = lane & 15;
    const int m0   = blockIdx.x * 128;

    for (int i = tid; i < 128 * 256; i += 256) {
        int k = i >> 8, n = i & 255;             // fw1 is [128][256]
        shW1[n * 132 + k] = f2bf(fw1[i]);
    }
    for (int i = tid; i < 256 * 128; i += 256) {
        int k = i >> 7, n = i & 127;             // fw2 is [256][128]
        shW2[n * 260 + k] = f2bf(fw2[i]);
    }
    for (int i = tid; i < 128 * 128; i += 256) {
        int mr = i >> 7, k = i & 127;
        int node = m0 + mr;
        float v = 0.f;
        if (node < nNodes) {
            float rv = r[(size_t)node * D_FEAT + k];
            v = g1[k] * (rv - mean1[k]) * rstd1[k] + bt1[k];
        }
        shH[mr * 132 + k] = f2bf(v);
    }
    if (tid < 128) { shSum[tid] = 0.f; shSumSq[tid] = 0.f; }
    __syncthreads();

    const v8f z8 = {0.f, 0.f, 0.f, 0.f, 0.f, 0.f, 0.f, 0.f};

    // GEMM1: F = relu(H @ FW1 + fb1), N=256 processed in two chunks of 128
    for (int ch = 0; ch < 2; ++ch) {
        v8f acc[8];
#pragma unroll
        for (int t = 0; t < 8; ++t) acc[t] = z8;
#pragma unroll
        for (int ks = 0; ks < 4; ++ks) {
            v16bf a = frag_a(shH + (wave * 16 + lm) * 132, half, ks);
#pragma unroll
            for (int t = 0; t < 8; ++t) {
                v16bf b = frag_b(shW1 + (ch * 128 + t * 16 + lm) * 132, half, ks);
                acc[t] = __builtin_amdgcn_wmma_f32_16x16x32_bf16(
                    false, a, false, b, (short)0, acc[t], false, false);
            }
        }
#pragma unroll
        for (int t = 0; t < 8; ++t) {
            int n = ch * 128 + t * 16 + lm;
            float bias = fb1[n];
#pragma unroll
            for (int i = 0; i < 8; ++i) {
                int m = wave * 16 + i + 8 * half;
                float v = fmaxf(acc[t][i] + bias, 0.f);
                shF[m * 260 + n] = f2bf(v);
            }
        }
        __syncthreads();
    }

    // GEMM2: out = BN1(r) + (F @ FW2 + fb2); K = 256 -> 8 k-steps
    {
        v8f acc[8];
#pragma unroll
        for (int t = 0; t < 8; ++t) acc[t] = z8;
#pragma unroll
        for (int ks = 0; ks < 8; ++ks) {
            v16bf a = frag_a(shF + (wave * 16 + lm) * 260, half, ks);
#pragma unroll
            for (int t = 0; t < 8; ++t) {
                v16bf b = frag_b(shW2 + (t * 16 + lm) * 260, half, ks);
                acc[t] = __builtin_amdgcn_wmma_f32_16x16x32_bf16(
                    false, a, false, b, (short)0, acc[t], false, false);
            }
        }
#pragma unroll
        for (int t = 0; t < 8; ++t) {
            int n = t * 16 + lm;
            float bias = fb2[n];
            float ls = 0.f, ls2 = 0.f;
#pragma unroll
            for (int i = 0; i < 8; ++i) {
                int m = wave * 16 + i + 8 * half;
                int node = m0 + m;
                if (node < nNodes) {
                    float rv = r[(size_t)node * D_FEAT + n];
                    float h  = g1[n] * (rv - mean1[n]) * rstd1[n] + bt1[n];
                    float sv = h + acc[t][i] + bias;
                    outp[(size_t)node * D_FEAT + n] = sv;
                    ls  += sv;
                    ls2 += sv * sv;
                }
            }
            ls  += __shfl_xor(ls, 16);
            ls2 += __shfl_xor(ls2, 16);
            if (half == 0) {
                atomicAdd(&shSum[n], ls);
                atomicAdd(&shSumSq[n], ls2);
            }
        }
    }
    __syncthreads();
    if (tid < 128) {
        atomicAdd(&gsum[tid], shSum[tid]);
        atomicAdd(&gsumsq[tid], shSumSq[tid]);
    }
}

// In-place BN2 on the output buffer.
__global__ void k_bn2(float* __restrict__ out,
                      const float* __restrict__ mean, const float* __restrict__ rstd,
                      const float* __restrict__ g, const float* __restrict__ b, size_t n) {
    size_t i = (size_t)blockIdx.x * blockDim.x + threadIdx.x;
    size_t stride = (size_t)gridDim.x * blockDim.x;
    for (; i < n; i += stride) {
        int c = (int)(i & 127);
        float v = out[i];
        out[i] = g[c] * (v - mean[c]) * rstd[c] + b[c];
    }
}

// ---------- launch ----------

extern "C" void kernel_launch(void* const* d_in, const int* in_sizes, int n_in,
                              void* d_out, int out_size, void* d_ws, size_t ws_size,
                              hipStream_t stream) {
    const float* x     = (const float*)d_in[0];
    const float* eattr = (const float*)d_in[1];
    const float* w1g   = (const float*)d_in[2];
    const float* b1g   = (const float*)d_in[3];
    const float* w2g   = (const float*)d_in[4];
    const float* b2g   = (const float*)d_in[5];
    const float* bn1g  = (const float*)d_in[6];
    const float* bn1b  = (const float*)d_in[7];
    const float* ffw1  = (const float*)d_in[8];
    const float* ffb1  = (const float*)d_in[9];
    const float* ffw2  = (const float*)d_in[10];
    const float* ffb2  = (const float*)d_in[11];
    const float* bn2g  = (const float*)d_in[12];
    const float* bn2b  = (const float*)d_in[13];
    const int*   eidx  = (const int*)d_in[14];
    (void)n_in; (void)out_size; (void)ws_size;
    (void)ffb1;

    const int nNodes = in_sizes[0] / D_FEAT;
    const int nEdges = in_sizes[1] / D_FEAT;

    float* agg   = (float*)d_ws;                        // N*128 f32; reused as r after k_gin
    float* stats = agg + (size_t)nNodes * D_FEAT;       // 1024 floats of stats
    float* sum1 = stats,        *sumsq1 = stats + 128;
    float* sum2 = stats + 256,  *sumsq2 = stats + 384;
    float* mean1 = stats + 512, *rstd1  = stats + 640;
    float* mean2 = stats + 768, *rstd2  = stats + 896;
    float* outp = (float*)d_out;

    // zero agg + the four stat-sum buffers (contiguous)
    k_zero<<<1024, 256, 0, stream>>>(agg, (size_t)nNodes * D_FEAT + 512);

    k_edges<<<4096, 256, 0, stream>>>(x, eattr, eidx, agg, nEdges);

    int nBlocks = (nNodes + 127) / 128;
    k_gin<<<nBlocks, 256, 0, stream>>>(x, agg, w1g, b1g, w2g, b2g,
                                       agg /*r aliases agg*/, sum1, sumsq1, nNodes);
    k_stats<<<1, 128, 0, stream>>>(sum1, sumsq1, mean1, rstd1, 1.0f / (float)nNodes);
    k_ffn<<<nBlocks, 256, 0, stream>>>(agg /*r*/, mean1, rstd1, bn1g, bn1b,
                                       ffw1, d_in[9] ? (const float*)d_in[9] : ffw1,
                                       ffw2, ffb2, outp, sum2, sumsq2, nNodes);
    k_stats<<<1, 128, 0, stream>>>(sum2, sumsq2, mean2, rstd2, 1.0f / (float)nNodes);
    k_bn2<<<2048, 256, 0, stream>>>(outp, mean2, rstd2, bn2g, bn2b, (size_t)nNodes * D_FEAT);
}
